// EncoderBlock_36833639530553
// MI455X (gfx1250) — matrix operand order repaired and checked by reading, not compiled, not run on hardware
//
#include <hip/hip_runtime.h>
#include <hip/hip_bf16.h>

// ---------------------------------------------------------------------------
// Encoder block for MI455X (gfx1250, wave32, WMMA).
// bf16 storage + v_wmma_f32_16x16x32_bf16 fp32-accumulate GEMMs.
// GEMM tile: 128x128x32 per 256-thread block (8 waves, wave = 64x32 = 8 C tiles),
// double-buffered LDS. Tile staging uses CDNA5 async global->LDS copies
// (GLOBAL_LOAD_ASYNC_TO_LDS_B128 / ASYNCcnt) when the toolchain exposes the
// builtin, else a register-staged prefetch pipeline.
// All B operands are [N x K] row-major (weights pre-transposed once; V stored
// transposed by the V-projection epilogue), so LDS staging is pure 128b copies.
// ---------------------------------------------------------------------------

typedef __attribute__((ext_vector_type(16))) __bf16     v16bf;
typedef __attribute__((ext_vector_type(8)))  float      v8f;
typedef __attribute__((ext_vector_type(4)))  unsigned   u32x4;

// exact parameter type expected by the async-copy builtin (from the clang
// diagnostic): GCC-style 4 x i32 vector, addrspace-qualified pointers
typedef __attribute__((__vector_size__(4 * sizeof(int)))) int i32x4v;
typedef __attribute__((address_space(1))) i32x4v g_i32x4;   // global
typedef __attribute__((address_space(3))) i32x4v l_i32x4;   // LDS

#if defined(__gfx1250__) && __has_builtin(__builtin_amdgcn_global_load_async_to_lds_b128)
#define ASYNC_COPY 1
#else
#define ASYNC_COPY 0
#endif

#if ASYNC_COPY
#if __has_builtin(__builtin_amdgcn_s_wait_asynccnt)
#define WAIT_ASYNC() __builtin_amdgcn_s_wait_asynccnt(0)
#else
#define WAIT_ASYNC() asm volatile("s_wait_asynccnt 0x0" ::: "memory")
#endif
__device__ __forceinline__ void async_cp16(const unsigned short* g, unsigned short* l) {
    __builtin_amdgcn_global_load_async_to_lds_b128(
        (g_i32x4*)(g),
        (l_i32x4*)(l),
        /*imm offset=*/0, /*cpol=*/0);
}
#endif

union Frag {            // 8 VGPRs viewed as 16 bf16 or 2x128b chunks
    v16bf v;
    u32x4 q[2];
};

__device__ __forceinline__ unsigned short f32_to_bf16(float f) {
    unsigned u = __float_as_uint(f);
    u += 0x7FFFu + ((u >> 16) & 1u);      // round to nearest even
    return (unsigned short)(u >> 16);
}

// ---------------------------------------------------------------------------
// fp32 -> bf16 convert (plain)
// ---------------------------------------------------------------------------
__global__ __launch_bounds__(256)
void cvt_f32_bf16(const float* __restrict__ in, unsigned short* __restrict__ out,
                  long long n) {
    long long i = (long long)blockIdx.x * blockDim.x + threadIdx.x;
    if (i < n) out[i] = f32_to_bf16(in[i]);
}

// ---------------------------------------------------------------------------
// fp32 [R x C] -> bf16 [C x R] tiled transpose-convert (coalesced both ways)
// ---------------------------------------------------------------------------
__global__ __launch_bounds__(256)
void cvtT_f32_bf16(const float* __restrict__ in, unsigned short* __restrict__ out,
                   int R, int C) {
    __shared__ unsigned short t[32][40];          // pad: conflict-free columns
    const int tx = threadIdx.x & 31;
    const int ty = threadIdx.x >> 5;              // 0..7
    const int r0 = blockIdx.y * 32, c0 = blockIdx.x * 32;
#pragma unroll
    for (int j = 0; j < 4; ++j)
        t[ty + j * 8][tx] = f32_to_bf16(in[(long long)(r0 + ty + j * 8) * C + c0 + tx]);
    __syncthreads();
#pragma unroll
    for (int j = 0; j < 4; ++j)
        out[(long long)(c0 + ty + j * 8) * R + r0 + tx] = t[tx][ty + j * 8];
}

// ---------------------------------------------------------------------------
// Batched bf16 WMMA GEMM:  C = alpha * A @ B^T + bias + residual
// A: [M x K] row-major bf16 (lda);  B: [N x K] row-major bf16 (ldb)
// out: fp32 (outF, ldc) and/or bf16 (outB); storeT!=0 stores outB transposed
// ([N x M], ldct) with one packed 16B store per C tile per lane.
// Batch z -> (z/H, z%H) with independent element strides.
// Requires M,N % 128 == 0 and K % 32 == 0 (true for all launches here).
// ---------------------------------------------------------------------------
#define BM 128
#define BN 128
#define BK 32
#define LDSS 40   // K-stride in LDS: 32 + 8 pad (80B rows, 16B aligned)

__global__ __launch_bounds__(256)
void gemm_bf16_wmma(const unsigned short* __restrict__ A,
                    const unsigned short* __restrict__ B,
                    const float* __restrict__ bias,
                    const float* __restrict__ residual,
                    float* __restrict__ outF,
                    unsigned short* __restrict__ outB,
                    int M, int N, int K,
                    int lda, int ldb, int ldc,
                    int storeT, int ldct, int relu, float alpha,
                    int H,
                    long long sAb, long long sAh,
                    long long sBb, long long sBh,
                    long long sCb, long long sCh) {
    __shared__ alignas(16) unsigned short As[2][BM * LDSS];   // 10 KB each
    __shared__ alignas(16) unsigned short Bs[2][BN * LDSS];   // 40 KB total

    const int tid  = threadIdx.x;
    const int lane = tid & 31;
    const int wave = tid >> 5;
    const int wm   = (wave & 1) * 64;     // wave M origin in block tile
    const int wn   = (wave >> 1) * 32;    // wave N origin in block tile
    const int l15  = lane & 15;
    const int lhi  = lane >> 4;           // lane half
    const int aklo = lhi * 8;             // A frag: e0..7 -> K=aklo.., e8..15 -> K=aklo+16..
    const int bklo = lhi * 8;             // B frag: e0..7 -> K=bklo.., e8..15 -> K=bklo+16..

    const int m0 = blockIdx.y * BM;
    const int n0 = blockIdx.x * BN;
    const int bz = blockIdx.z;
    const long long offA = (long long)(bz / H) * sAb + (long long)(bz % H) * sAh;
    const long long offB = (long long)(bz / H) * sBb + (long long)(bz % H) * sBh;
    const long long offC = (long long)(bz / H) * sCb + (long long)(bz % H) * sCh;

    const unsigned short* Ab = A + offA;
    const unsigned short* Bb = B + offB;

    // per-thread staging geometry: 2 x 16B chunks per matrix per tile
    const int srow = tid >> 2;            // +64 rows for the second chunk
    const int scc  = (tid & 3) * 8;

    v8f acc[4][2];
#pragma unroll
    for (int mt = 0; mt < 4; ++mt)
#pragma unroll
        for (int nt = 0; nt < 2; ++nt)
#pragma unroll
            for (int r = 0; r < 8; ++r) acc[mt][nt][r] = 0.0f;

    const int nk = K / BK;
    int cur = 0;

#if ASYNC_COPY
    // ---- async global->LDS pipeline (no VGPR staging, no ds_store drain) ----
    auto issue_tile = [&](int buf, int kofs) {
#pragma unroll
        for (int i = 0; i < 2; ++i) {
            const int row = srow + i * 64;
            async_cp16(Ab + (long long)(m0 + row) * lda + kofs + scc,
                       &As[buf][row * LDSS + scc]);
            async_cp16(Bb + (long long)(n0 + row) * ldb + kofs + scc,
                       &Bs[buf][row * LDSS + scc]);
        }
    };
    issue_tile(0, 0);
    for (int it = 0; it < nk; ++it) {
        WAIT_ASYNC();                     // this wave's copies into buf[cur] landed
        __syncthreads();                  // all waves' copies visible; buf[cur^1] drained
        if (it + 1 < nk) issue_tile(cur ^ 1, (it + 1) * BK);
#else
    // ---- register-staged prefetch pipeline (fallback) ----
    u32x4 ra[2], rb[2];
#pragma unroll
    for (int i = 0; i < 2; ++i) {
        const int row = srow + i * 64;
        ra[i] = *(const u32x4*)(Ab + (long long)(m0 + row) * lda + scc);
        rb[i] = *(const u32x4*)(Bb + (long long)(n0 + row) * ldb + scc);
    }
#pragma unroll
    for (int i = 0; i < 2; ++i) {
        const int row = srow + i * 64;
        *(u32x4*)&As[0][row * LDSS + scc] = ra[i];
        *(u32x4*)&Bs[0][row * LDSS + scc] = rb[i];
    }
    for (int it = 0; it < nk; ++it) {
        __syncthreads();
        if (it + 1 < nk) {
            const int kn = (it + 1) * BK;
#pragma unroll
            for (int i = 0; i < 2; ++i) {
                const int row = srow + i * 64;
                ra[i] = *(const u32x4*)(Ab + (long long)(m0 + row) * lda + kn + scc);
                rb[i] = *(const u32x4*)(Bb + (long long)(n0 + row) * ldb + kn + scc);
            }
        }
#endif
        // ---- compute on buf[cur]: 12 ds_load_b128 -> 8 WMMAs ----
        Frag af[4], bfr[2];
#pragma unroll
        for (int mt = 0; mt < 4; ++mt) {
            const unsigned short* p = &As[cur][(wm + mt * 16 + l15) * LDSS + aklo];
            af[mt].q[0] = *(const u32x4*)(p);
            af[mt].q[1] = *(const u32x4*)(p + 16);
        }
#pragma unroll
        for (int nt = 0; nt < 2; ++nt) {
            const unsigned short* p = &Bs[cur][(wn + nt * 16 + l15) * LDSS + bklo];
            bfr[nt].q[0] = *(const u32x4*)(p);
            bfr[nt].q[1] = *(const u32x4*)(p + 16);
        }
#pragma unroll
        for (int mt = 0; mt < 4; ++mt)
#pragma unroll
            for (int nt = 0; nt < 2; ++nt)
                acc[mt][nt] = __builtin_amdgcn_wmma_f32_16x16x32_bf16(
                    false, af[mt].v, false, bfr[nt].v,
                    (short)0, acc[mt][nt], false, false);
#if !ASYNC_COPY
        if (it + 1 < nk) {
            const int nxt = cur ^ 1;
#pragma unroll
            for (int i = 0; i < 2; ++i) {
                const int row = srow + i * 64;
                *(u32x4*)&As[nxt][row * LDSS + scc] = ra[i];
                *(u32x4*)&Bs[nxt][row * LDSS + scc] = rb[i];
            }
        }
#endif
        cur ^= 1;
    }

    // ---- epilogue: alpha, bias, residual, relu, fp32/bf16/bf16-transposed ----
#pragma unroll
    for (int mt = 0; mt < 4; ++mt) {
#pragma unroll
        for (int nt = 0; nt < 2; ++nt) {
            const int col   = n0 + wn + nt * 16 + l15;
            const float bv  = bias ? bias[col] : 0.0f;
            const int rbase = m0 + wm + mt * 16 + lhi * 8;
            float vals[8];
#pragma unroll
            for (int r = 0; r < 8; ++r) {
                float val = acc[mt][nt][r] * alpha + bv;
                if (residual) val += residual[offC + (long long)(rbase + r) * ldc + col];
                if (relu) val = fmaxf(val, 0.0f);
                vals[r] = val;
            }
            if (outF) {
#pragma unroll
                for (int r = 0; r < 8; ++r)
                    outF[offC + (long long)(rbase + r) * ldc + col] = vals[r];
            }
            if (outB) {
                if (storeT) {   // lane's 8 rows are contiguous in C^T: 1x16B store
                    union { u32x4 q; unsigned short u[8]; } pk;
#pragma unroll
                    for (int r = 0; r < 8; ++r) pk.u[r] = f32_to_bf16(vals[r]);
                    *(u32x4*)&outB[(long long)col * ldct + rbase] = pk.q;
                } else {
#pragma unroll
                    for (int r = 0; r < 8; ++r)
                        outB[offC + (long long)(rbase + r) * ldc + col] =
                            f32_to_bf16(vals[r]);
                }
            }
        }
    }
}

// ---------------------------------------------------------------------------
// Row softmax: one 256-thread block per row of 1024 fp32 scores -> bf16 probs
// ---------------------------------------------------------------------------
__global__ __launch_bounds__(256)
void softmax1024(const float* __restrict__ S, unsigned short* __restrict__ P) {
    const long long row = blockIdx.x;
    const int tid = threadIdx.x;
    const float* sr = S + row * 1024;
    __shared__ float red[256];

    float v[4], m = -3.4e38f;
#pragma unroll
    for (int i = 0; i < 4; ++i) { v[i] = sr[tid + i * 256]; m = fmaxf(m, v[i]); }
    red[tid] = m; __syncthreads();
    for (int o = 128; o > 0; o >>= 1) {
        if (tid < o) red[tid] = fmaxf(red[tid], red[tid + o]);
        __syncthreads();
    }
    m = red[0]; __syncthreads();

    float s = 0.0f;
#pragma unroll
    for (int i = 0; i < 4; ++i) { v[i] = __expf(v[i] - m); s += v[i]; }
    red[tid] = s; __syncthreads();
    for (int o = 128; o > 0; o >>= 1) {
        if (tid < o) red[tid] += red[tid + o];
        __syncthreads();
    }
    const float inv = 1.0f / red[0];
#pragma unroll
    for (int i = 0; i < 4; ++i)
        P[row * 1024 + tid + i * 256] = f32_to_bf16(v[i] * inv);
}

// ---------------------------------------------------------------------------
// Host-side orchestration
// ---------------------------------------------------------------------------
extern "C" void kernel_launch(void* const* d_in, const int* in_sizes, int n_in,
                              void* d_out, int out_size, void* d_ws, size_t ws_size,
                              hipStream_t stream) {
    (void)in_sizes; (void)n_in; (void)out_size; (void)ws_size;

    const float* x   = (const float*)d_in[0];
    const float* wq  = (const float*)d_in[1];
    const float* bq  = (const float*)d_in[2];
    const float* wk  = (const float*)d_in[3];
    const float* bk  = (const float*)d_in[4];
    const float* wv  = (const float*)d_in[5];
    const float* bv  = (const float*)d_in[6];
    const float* wfc = (const float*)d_in[7];
    const float* bfc = (const float*)d_in[8];
    const float* w1  = (const float*)d_in[9];
    const float* b1  = (const float*)d_in[10];
    const float* w2  = (const float*)d_in[11];
    const float* b2  = (const float*)d_in[12];
    float* out = (float*)d_out;

    const long long T  = 4096;   // b*s tokens
    const long long D  = 1024;   // d_model
    const long long HW = 8192;   // h*head_dim
    const long long DH = 4096;   // ffn hidden
    const long long S  = 1024;   // seq len
    const int BH = 32;           // b*h attention batches

    // workspace carve-out (256B aligned)
    char* ws = (char*)d_ws;
    size_t off = 0;
    auto carve = [&](size_t bytes) -> void* {
        void* p = ws + off;
        off += (bytes + 255) & ~(size_t)255;
        return p;
    };
    unsigned short* xb   = (unsigned short*)carve(T * D * 2);
    unsigned short* wqT  = (unsigned short*)carve(D * HW * 2);   // [HW x D]
    unsigned short* wkT  = (unsigned short*)carve(D * HW * 2);   // [HW x D]
    unsigned short* wvT  = (unsigned short*)carve(D * HW * 2);   // [HW x D]
    unsigned short* wfcT = (unsigned short*)carve(HW * D * 2);   // [D x HW]
    unsigned short* w1T  = (unsigned short*)carve(D * DH * 2);   // [DH x D]
    unsigned short* w2T  = (unsigned short*)carve(DH * D * 2);   // [D x DH]
    unsigned short* Qb   = (unsigned short*)carve(T * HW * 2);   // [T x HW]
    unsigned short* Kb   = (unsigned short*)carve(T * HW * 2);   // [T x HW]
    unsigned short* Vt   = (unsigned short*)carve(T * HW * 2);   // [HW x T]
    float*          sc   = (float*)carve((size_t)BH * S * S * 4);
    unsigned short* pb   = (unsigned short*)carve((size_t)BH * S * S * 2);
    unsigned short* Zb   = (unsigned short*)carve(T * HW * 2);   // [T x HW]
    float*          zf   = (float*)carve(T * D * 4);
    unsigned short* zb2  = (unsigned short*)carve(T * D * 2);
    unsigned short* hb   = (unsigned short*)carve(T * DH * 2);

    // ---- one-time conversions ----
    cvt_f32_bf16<<<dim3((unsigned)((T * D + 255) / 256)), dim3(256), 0, stream>>>(
        x, xb, T * D);
    auto cvtT = [&](const float* in, unsigned short* o, int R, int C) {
        cvtT_f32_bf16<<<dim3(C / 32, R / 32), dim3(256), 0, stream>>>(in, o, R, C);
    };
    cvtT(wq, wqT, (int)D, (int)HW);     // -> [HW x D]
    cvtT(wk, wkT, (int)D, (int)HW);
    cvtT(wv, wvT, (int)D, (int)HW);
    cvtT(wfc, wfcT, (int)HW, (int)D);   // -> [D x HW]
    cvtT(w1, w1T, (int)D, (int)DH);     // -> [DH x D]
    cvtT(w2, w2T, (int)DH, (int)D);     // -> [D x DH]

    auto gemm = [&](const unsigned short* A, const unsigned short* B,
                    const float* bias, const float* res,
                    float* oF, unsigned short* oB,
                    int M, int N, int K, int lda, int ldb, int ldc,
                    int storeT, int ldct, int relu, float alpha,
                    int batch, int H,
                    long long sAb, long long sAh,
                    long long sBb, long long sBh,
                    long long sCb, long long sCh) {
        dim3 grid(N / BN, M / BM, batch);
        gemm_bf16_wmma<<<grid, dim3(256), 0, stream>>>(
            A, B, bias, res, oF, oB, M, N, K, lda, ldb, ldc,
            storeT, ldct, relu, alpha, H, sAb, sAh, sBb, sBh, sCb, sCh);
    };

    // Q/K projections: [4096x1024] @ wT^T -> bf16 [T x HW]
    gemm(xb, wqT, bq, nullptr, nullptr, Qb, 4096, 8192, 1024, 1024, 1024, 8192,
         0, 0, 0, 1.0f, 1, 1, 0, 0, 0, 0, 0, 0);
    gemm(xb, wkT, bk, nullptr, nullptr, Kb, 4096, 8192, 1024, 1024, 1024, 8192,
         0, 0, 0, 1.0f, 1, 1, 0, 0, 0, 0, 0, 0);
    // V projection: epilogue stores V transposed -> Vt [HW x T]
    gemm(xb, wvT, bv, nullptr, nullptr, Vt, 4096, 8192, 1024, 1024, 1024, 8192,
         1, (int)T, 0, 1.0f, 1, 1, 0, 0, 0, 0, 0, 0);

    // scores[bh] = (Q_bh @ K_bh^T) / 32, batched over 32 (b,h)
    gemm(Qb, Kb, nullptr, nullptr, sc, nullptr, 1024, 1024, 1024, 8192, 8192, 1024,
         0, 0, 0, 1.0f / 32.0f, BH, 8,
         S * HW, D,            // A strides into Qb
         S * HW, D,            // B strides into Kb
         8LL * S * S, S * S);  // C strides into scores

    softmax1024<<<dim3((unsigned)(BH * S)), dim3(256), 0, stream>>>(sc, pb);

    // Z[bh] = P_bh @ (Vt_bh)^T, into [T x HW] head-concat layout
    gemm(pb, Vt, nullptr, nullptr, nullptr, Zb, 1024, 1024, 1024, 1024, 4096, 8192,
         0, 0, 0, 1.0f, BH, 8,
         8LL * S * S, S * S,   // A strides into probs
         S, D * T,             // B strides into Vt: +b*1024 cols, +h*1024 rows
         S * HW, D);           // C strides into Zb

    // attn_out + residual1: z = Z @ wfc + bfc + x (fp32 + bf16)
    gemm(Zb, wfcT, bfc, x, zf, zb2, 4096, 1024, 8192, 8192, 8192, 1024,
         0, 0, 0, 1.0f, 1, 1, 0, 0, 0, 0, 0, 0);

    // hidden = relu(z @ w1 + b1) -> bf16
    gemm(zb2, w1T, b1, nullptr, nullptr, hb, 4096, 4096, 1024, 1024, 1024, 4096,
         0, 0, 1, 1.0f, 1, 1, 0, 0, 0, 0, 0, 0);

    // out = hidden @ w2 + b2 + z
    gemm(hb, w2T, b2, zf, out, nullptr, 4096, 1024, 4096, 4096, 4096, 1024,
         0, 0, 0, 1.0f, 1, 1, 0, 0, 0, 0, 0, 0);
}
